// FlaxDbrxNormAttentionNorm_29472065585698
// MI455X (gfx1250) — compile-verified, matrix operand order checked by reading, not run on hardware
//
#include <hip/hip_runtime.h>
#include <stdint.h>

// ---------------------------------------------------------------- types ----
typedef __attribute__((ext_vector_type(16))) __bf16 bf16x16;
typedef __attribute__((ext_vector_type(8)))  float  f32x8;
typedef __attribute__((ext_vector_type(4)))  unsigned int u32x4;

union FragU { bf16x16 v; u32x4 q[2]; };

static constexpr int Bb   = 2;
static constexpr int Ss   = 2048;
static constexpr int Dd   = 2048;
static constexpr int Hh   = 32;
static constexpr int KVH  = 8;
static constexpr int HD   = 64;
static constexpr int NQKV = Dd + 2 * KVH * HD;   // 3072
static constexpr int Mrows = Bb * Ss;            // 4096

// float -> bf16, round-to-nearest-even
__device__ __forceinline__ unsigned short f2bf(float f) {
    unsigned int x = __float_as_uint(f);
    unsigned int r = x + 0x7FFFu + ((x >> 16) & 1u);
    return (unsigned short)(r >> 16);
}

// Load one 16x32 fragment given a per-lane base that already includes
// row*ld + half*8: two contiguous 16B chunks at +0 and +16 elements.
__device__ __forceinline__ bf16x16 load_chunk(const unsigned short* __restrict__ p) {
    FragU u;
    u.q[0] = *(const u32x4*)(p);
    u.q[1] = *(const u32x4*)(p + 16);
    return u.v;
}

// Load a 16x32 bf16 A/B fragment (K = half*8 + {0..7} and 16+half*8+{0..7}).
__device__ __forceinline__ bf16x16 load_frag16(const unsigned short* __restrict__ p,
                                               int ld, int row, int k0, int half) {
    return load_chunk(p + (size_t)row * ld + k0 + half * 8);
}

__device__ __forceinline__ f32x8 wmma_bf16(bf16x16 a, bf16x16 b, f32x8 c) {
    return __builtin_amdgcn_wmma_f32_16x16x32_bf16(false, a, false, b,
                                                   (short)0, c, false, false);
}

// ------------------------------------------------- transpose + f32->bf16 ---
// src [R, C] f32 -> dst [C, R] bf16
__global__ __launch_bounds__(256) void transpose_bf16(const float* __restrict__ src,
                                                      unsigned short* __restrict__ dst,
                                                      int R, int C) {
    int c = blockIdx.x * 32 + (threadIdx.x & 31);
    int r = blockIdx.y * 8 + (threadIdx.x >> 5);
    if (c < C && r < R)
        dst[(size_t)c * R + r] = f2bf(src[(size_t)r * C + c]);
}

// ------------------------------------------------------------- layernorm ---
__device__ __forceinline__ float block_sum(float v, float* sbuf) {
#pragma unroll
    for (int m = 16; m >= 1; m >>= 1) v += __shfl_xor(v, m, 32);
    int w = threadIdx.x >> 5, l = threadIdx.x & 31;
    __syncthreads();
    if (l == 0) sbuf[w] = v;
    __syncthreads();
    float s = sbuf[0];
#pragma unroll
    for (int i = 1; i < 8; i++) s += sbuf[i];
    return s;
}

template <bool BF16OUT>
__global__ __launch_bounds__(256) void layernorm_kern(const float* __restrict__ x,
                                                      const float* __restrict__ g,
                                                      void* __restrict__ yout) {
    __shared__ float sbuf[8];
    int row = blockIdx.x;
    const float* xr = x + (size_t)row * Dd;
    float vals[8];
    float s = 0.f;
#pragma unroll
    for (int i = 0; i < 8; i++) { vals[i] = xr[threadIdx.x + i * 256]; s += vals[i]; }
    float mean = block_sum(s, sbuf) * (1.0f / (float)Dd);
    float vs = 0.f;
#pragma unroll
    for (int i = 0; i < 8; i++) { float d = vals[i] - mean; vs += d * d; }
    float var = block_sum(vs, sbuf) * (1.0f / (float)Dd);
    float rstd = rsqrtf(var + 1e-6f);
#pragma unroll
    for (int i = 0; i < 8; i++) {
        int c = threadIdx.x + i * 256;
        float o = (vals[i] - mean) * rstd * g[c];
        if (BF16OUT) ((unsigned short*)yout)[(size_t)row * Dd + c] = f2bf(o);
        else         ((float*)yout)[(size_t)row * Dd + c] = o;
    }
}

// ------------------------------------------------------------------ GEMM ---
// C[M,N] = A[M,K](bf16, row-major) x Bt[N,K](bf16, row-major == B^T)
// Block: 256 threads = 8 waves; each wave: 32(M) x 64(N), 8 accumulators,
// software-pipelined fragment loads (next k-step issued before current WMMAs).
__global__ __launch_bounds__(256) void gemm_bf16(const unsigned short* __restrict__ A,
                                                 const unsigned short* __restrict__ Bt,
                                                 float* __restrict__ C,
                                                 int M, int N, int K,
                                                 int doClip, float clipV,
                                                 const float* __restrict__ resid,
                                                 int addResid) {
    int wave = threadIdx.x >> 5;
    int lane = threadIdx.x & 31;
    int half = lane >> 4, ln16 = lane & 15;
    int mTile = blockIdx.x * 256 + wave * 32;
    int nBase = blockIdx.y * 64;

    const unsigned short* pa0 = A + (size_t)(mTile + ln16) * K + half * 8;
    const unsigned short* pa1 = A + (size_t)(mTile + 16 + ln16) * K + half * 8;
    const unsigned short* pb0 = Bt + (size_t)(nBase + ln16) * K + half * 8;
    const unsigned short* pb1 = Bt + (size_t)(nBase + 16 + ln16) * K + half * 8;
    const unsigned short* pb2 = Bt + (size_t)(nBase + 32 + ln16) * K + half * 8;
    const unsigned short* pb3 = Bt + (size_t)(nBase + 48 + ln16) * K + half * 8;

    f32x8 zero = {0.f, 0.f, 0.f, 0.f, 0.f, 0.f, 0.f, 0.f};
    f32x8 acc0[4], acc1[4];
#pragma unroll
    for (int t = 0; t < 4; t++) { acc0[t] = zero; acc1[t] = zero; }

    // prologue: k-step 0 fragments
    bf16x16 a0 = load_chunk(pa0);
    bf16x16 a1 = load_chunk(pa1);
    bf16x16 b0 = load_chunk(pb0);
    bf16x16 b1 = load_chunk(pb1);
    bf16x16 b2 = load_chunk(pb2);
    bf16x16 b3 = load_chunk(pb3);

    for (int k0 = 32; k0 < K; k0 += 32) {
        // issue next k-step loads first (overlap with WMMAs below)
        bf16x16 na0 = load_chunk(pa0 + k0);
        bf16x16 na1 = load_chunk(pa1 + k0);
        bf16x16 nb0 = load_chunk(pb0 + k0);
        bf16x16 nb1 = load_chunk(pb1 + k0);
        bf16x16 nb2 = load_chunk(pb2 + k0);
        bf16x16 nb3 = load_chunk(pb3 + k0);
        // prefetch A two k-steps ahead (speculative; OOB is dropped)
        __builtin_prefetch((const void*)(pa0 + k0 + 32), 0, 1);
        __builtin_prefetch((const void*)(pa1 + k0 + 32), 0, 1);

        acc0[0] = wmma_bf16(a0, b0, acc0[0]);
        acc1[0] = wmma_bf16(a1, b0, acc1[0]);
        acc0[1] = wmma_bf16(a0, b1, acc0[1]);
        acc1[1] = wmma_bf16(a1, b1, acc1[1]);
        acc0[2] = wmma_bf16(a0, b2, acc0[2]);
        acc1[2] = wmma_bf16(a1, b2, acc1[2]);
        acc0[3] = wmma_bf16(a0, b3, acc0[3]);
        acc1[3] = wmma_bf16(a1, b3, acc1[3]);

        a0 = na0; a1 = na1; b0 = nb0; b1 = nb1; b2 = nb2; b3 = nb3;
    }
    // epilogue k-step
    acc0[0] = wmma_bf16(a0, b0, acc0[0]);
    acc1[0] = wmma_bf16(a1, b0, acc1[0]);
    acc0[1] = wmma_bf16(a0, b1, acc0[1]);
    acc1[1] = wmma_bf16(a1, b1, acc1[1]);
    acc0[2] = wmma_bf16(a0, b2, acc0[2]);
    acc1[2] = wmma_bf16(a1, b2, acc1[2]);
    acc0[3] = wmma_bf16(a0, b3, acc0[3]);
    acc1[3] = wmma_bf16(a1, b3, acc1[3]);

#pragma unroll
    for (int t = 0; t < 4; t++) {
        int n = nBase + t * 16 + ln16;
#pragma unroll
        for (int r = 0; r < 8; r++) {
            int m0 = mTile + half * 8 + r;
            int m1 = m0 + 16;
            float v0 = acc0[t][r];
            float v1 = acc1[t][r];
            if (doClip) {
                v0 = fminf(fmaxf(v0, -clipV), clipV);
                v1 = fminf(fmaxf(v1, -clipV), clipV);
            }
            if (addResid) {
                v0 += resid[(size_t)m0 * N + n];
                v1 += resid[(size_t)m1 * N + n];
            }
            C[(size_t)m0 * N + n] = v0;
            C[(size_t)m1 * N + n] = v1;
        }
    }
}

// ------------------------------------------------------------ RoPE + pack --
// qkv f32 [B*S, 3072] -> Qs bf16 [B,H,S,HD] (x SCALE), Ks bf16 [B,KVH,S,HD],
// Vt bf16 [B,KVH,HD,S]
__global__ __launch_bounds__(256) void rope_pack(const float* __restrict__ qkv,
                                                 const int* __restrict__ pos,
                                                 unsigned short* __restrict__ Qs,
                                                 unsigned short* __restrict__ Ks,
                                                 unsigned short* __restrict__ Vt) {
    int bs = blockIdx.x;
    int b = bs >> 11, s = bs & (Ss - 1);
    float p = (float)pos[bs];
    const float* row = qkv + (size_t)bs * NQKV;
#pragma unroll
    for (int e = 0; e < 12; e++) {
        int idx = threadIdx.x + e * 256;
        float xv = row[idx];
        if (idx < Dd) {                          // Q
            int h = idx >> 6, d = idx & 63, j = d & 31;
            float inv = __powf(500000.0f, -(float)j * (1.0f / 32.0f));
            float sn, cs; __sincosf(p * inv, &sn, &cs);
            float partner = row[(d < 32) ? idx + 32 : idx - 32];
            float o = (d < 32) ? (xv * cs - partner * sn) : (xv * cs + partner * sn);
            o *= 0.125f;                         // 1/sqrt(64)
            Qs[(((size_t)b * Hh + h) * Ss + s) * HD + d] = f2bf(o);
        } else if (idx < Dd + KVH * HD) {        // K
            int t = idx - Dd;
            int h = t >> 6, d = t & 63, j = d & 31;
            float inv = __powf(500000.0f, -(float)j * (1.0f / 32.0f));
            float sn, cs; __sincosf(p * inv, &sn, &cs);
            float partner = row[(d < 32) ? idx + 32 : idx - 32];
            float o = (d < 32) ? (xv * cs - partner * sn) : (xv * cs + partner * sn);
            Ks[(((size_t)b * KVH + h) * Ss + s) * HD + d] = f2bf(o);
        } else {                                 // V (transposed)
            int t = idx - (Dd + KVH * HD);
            int h = t >> 6, d = t & 63;
            Vt[(((size_t)b * KVH + h) * HD + d) * Ss + s] = f2bf(xv);
        }
    }
}

// -------------------------------------------------------- flash attention --
// 1 wave per (b, h, 16-query tile). Online softmax; P staged via LDS.
__global__ __launch_bounds__(32) void attention(const unsigned short* __restrict__ Qs,
                                                const unsigned short* __restrict__ Ks,
                                                const unsigned short* __restrict__ Vt,
                                                const int* __restrict__ amask,
                                                unsigned short* __restrict__ attn) {
    __shared__ __align__(16) unsigned short Plds[16 * 32];
    int lane = threadIdx.x & 31;
    int half = lane >> 4, ln16 = lane & 15;
    int qTile = blockIdx.x * 16;
    int h = blockIdx.y, b = blockIdx.z;
    int kvh = h >> 2;

    const unsigned short* Qh = Qs + (((size_t)b * Hh + h) * Ss) * HD;
    const unsigned short* Kh = Ks + (((size_t)b * KVH + kvh) * Ss) * HD;
    const unsigned short* Vh = Vt + (((size_t)b * KVH + kvh) * HD) * Ss;

    bf16x16 aq0 = load_frag16(Qh, HD, qTile + ln16, 0, half);
    bf16x16 aq1 = load_frag16(Qh, HD, qTile + ln16, 32, half);

    f32x8 zero = {0.f, 0.f, 0.f, 0.f, 0.f, 0.f, 0.f, 0.f};
    f32x8 accO[4];
#pragma unroll
    for (int t = 0; t < 4; t++) accO[t] = zero;
    float mrow[8], lrow[8];
#pragma unroll
    for (int r = 0; r < 8; r++) { mrow[r] = -1e30f; lrow[r] = 0.f; }

    for (int kb = 0; kb < qTile + 16; kb += 32) {
        f32x8 sc0 = zero, sc1 = zero;
        {
            bf16x16 bk0 = load_frag16(Kh, HD, kb + ln16, 0, half);
            bf16x16 bk1 = load_frag16(Kh, HD, kb + ln16, 32, half);
            bf16x16 bk2 = load_frag16(Kh, HD, kb + 16 + ln16, 0, half);
            bf16x16 bk3 = load_frag16(Kh, HD, kb + 16 + ln16, 32, half);
            sc0 = wmma_bf16(aq0, bk0, sc0);
            sc0 = wmma_bf16(aq1, bk1, sc0);
            sc1 = wmma_bf16(aq0, bk2, sc1);
            sc1 = wmma_bf16(aq1, bk3, sc1);
        }
        int col0 = kb + ln16, col1 = kb + 16 + ln16;
        int am0 = amask[b * Ss + col0];
        int am1 = amask[b * Ss + col1];
        float al[8], p0[8], p1[8];
#pragma unroll
        for (int r = 0; r < 8; r++) {
            int rowq = qTile + half * 8 + r;
            float s0 = (col0 > rowq || am0 == 0) ? -1e30f : sc0[r];
            float s1 = (col1 > rowq || am1 == 0) ? -1e30f : sc1[r];
            float t = fmaxf(s0, s1);
            t = fmaxf(t, __shfl_xor(t, 1, 32));
            t = fmaxf(t, __shfl_xor(t, 2, 32));
            t = fmaxf(t, __shfl_xor(t, 4, 32));
            t = fmaxf(t, __shfl_xor(t, 8, 32));
            float nm = fmaxf(mrow[r], t);
            al[r] = __expf(mrow[r] - nm);
            p0[r] = __expf(s0 - nm);
            p1[r] = __expf(s1 - nm);
            float sm = p0[r] + p1[r];
            sm += __shfl_xor(sm, 1, 32);
            sm += __shfl_xor(sm, 2, 32);
            sm += __shfl_xor(sm, 4, 32);
            sm += __shfl_xor(sm, 8, 32);
            lrow[r] = lrow[r] * al[r] + sm;
            mrow[r] = nm;
        }
#pragma unroll
        for (int t = 0; t < 4; t++)
#pragma unroll
            for (int r = 0; r < 8; r++) accO[t][r] *= al[r];
#pragma unroll
        for (int r = 0; r < 8; r++) {
            Plds[(half * 8 + r) * 32 + ln16]      = f2bf(p0[r]);
            Plds[(half * 8 + r) * 32 + 16 + ln16] = f2bf(p1[r]);
        }
        __syncthreads();
        FragU pu;
        pu.q[0] = *(const u32x4*)(Plds + ln16 * 32 + half * 8);
        pu.q[1] = *(const u32x4*)(Plds + ln16 * 32 + half * 8 + 16);
#pragma unroll
        for (int t = 0; t < 4; t++) {
            bf16x16 bv = load_frag16(Vh, Ss, t * 16 + ln16, kb, half);
            accO[t] = wmma_bf16(pu.v, bv, accO[t]);
        }
        __syncthreads();
    }
#pragma unroll
    for (int t = 0; t < 4; t++) {
#pragma unroll
        for (int r = 0; r < 8; r++) {
            int rowq = qTile + half * 8 + r;
            float o = accO[t][r] / lrow[r];
            attn[((size_t)(b * Ss + rowq)) * Dd + h * HD + t * 16 + ln16] = f2bf(o);
        }
    }
}

// ---------------------------------------------------------------- launch ---
extern "C" void kernel_launch(void* const* d_in, const int* in_sizes, int n_in,
                              void* d_out, int out_size, void* d_ws, size_t ws_size,
                              hipStream_t stream) {
    const float* hidden   = (const float*)d_in[0];
    const int*   attnmask = (const int*)d_in[1];
    const int*   posids   = (const int*)d_in[2];
    // d_in[3] = causal mask (recomputed analytically)
    const float* ln1s     = (const float*)d_in[4];
    const float* w_qkv    = (const float*)d_in[5];
    const float* w_out    = (const float*)d_in[6];
    const float* ln2s     = (const float*)d_in[7];

    char* ws = (char*)d_ws;
    size_t off = 0;
    unsigned short* xln   = (unsigned short*)(ws + off); off += (size_t)Mrows * Dd * 2;
    unsigned short* wqkvT = (unsigned short*)(ws + off); off += (size_t)NQKV * Dd * 2;
    unsigned short* woutT = (unsigned short*)(ws + off); off += (size_t)Dd * Dd * 2;
    float*          qkv   = (float*)(ws + off);          off += (size_t)Mrows * NQKV * 4;
    unsigned short* Qsb   = (unsigned short*)(ws + off); off += (size_t)Bb * Hh * Ss * HD * 2;
    unsigned short* Ksb   = (unsigned short*)(ws + off); off += (size_t)Bb * KVH * Ss * HD * 2;
    unsigned short* Vtb   = (unsigned short*)(ws + off); off += (size_t)Bb * KVH * HD * Ss * 2;
    unsigned short* attnb = (unsigned short*)(ws + off); off += (size_t)Mrows * Dd * 2;

    float* out_resid = (float*)d_out;                      // residual2
    float* out_ln    = (float*)d_out + (size_t)Mrows * Dd; // layernorm2 output

    // 1) weight transpose+convert
    transpose_bf16<<<dim3((NQKV + 31) / 32, (Dd + 7) / 8), 256, 0, stream>>>(w_qkv, wqkvT, Dd, NQKV);
    transpose_bf16<<<dim3((Dd + 31) / 32, (Dd + 7) / 8), 256, 0, stream>>>(w_out, woutT, Dd, Dd);
    // 2) LN1 -> bf16
    layernorm_kern<true><<<Mrows, 256, 0, stream>>>(hidden, ln1s, (void*)xln);
    // 3) QKV GEMM (+clip)
    gemm_bf16<<<dim3(Mrows / 256, NQKV / 64), 256, 0, stream>>>(
        xln, wqkvT, qkv, Mrows, NQKV, Dd, 1, 8.0f, nullptr, 0);
    // 4) RoPE + pack
    rope_pack<<<Mrows, 256, 0, stream>>>(qkv, posids, Qsb, Ksb, Vtb);
    // 5) attention
    attention<<<dim3(Ss / 16, Hh, Bb), 32, 0, stream>>>(Qsb, Ksb, Vtb, attnmask, attnb);
    // 6) out-proj + residual -> residual2
    gemm_bf16<<<dim3(Mrows / 256, Dd / 64), 256, 0, stream>>>(
        attnb, woutT, out_resid, Mrows, Dd, Dd, 0, 0.0f, hidden, 1);
    // 7) LN2 -> f32
    layernorm_kern<false><<<Mrows, 256, 0, stream>>>(out_resid, ln2s, (void*)out_ln);

    (void)in_sizes; (void)n_in; (void)out_size; (void)ws_size;
}
